// AttentionPromptExtrapolation_56719338111252
// MI455X (gfx1250) — matrix operand order repaired and checked by reading, not compiled, not run on hardware
//
#include <hip/hip_runtime.h>

typedef __attribute__((ext_vector_type(16))) _Float16 v16h;
typedef __attribute__((ext_vector_type(8)))  _Float16 v8h;
typedef __attribute__((ext_vector_type(8)))  float    v8f;
typedef __attribute__((ext_vector_type(4)))  float    v4f;

#define D_       128
#define K_       64            // prompts per table
#define N_       512
#define T_MTI_   24
#define M_TOTAL  409600        // 32*512*25 rows
#define NTILES   (M_TOTAL/16)  // 25600
#define LDW      136           // padded row (halves): conflict-free ds_load_b128
#define WAVES    4
#define THREADS  (WAVES*32)

__global__ __launch_bounds__(THREADS)
void ape_kernel(const float* __restrict__ patches,
                const float* __restrict__ uP,
                const float* __restrict__ mP,
                const int*   __restrict__ s_mti,
                float*       __restrict__ out)
{
    // Combined transposed weights: sWt[d*LDW + kk] = W''[kk][d],
    // where W'' rows 0..63 = U, rows 64..127 = M.
    __shared__ _Float16 sWt[D_ * LDW];
    __shared__ _Float16 sS[WAVES][16 * LDW];   // per-wave masked scores S'' (16 x 128)
    __shared__ unsigned char snm[N_];          // 0 where n in s_mti

    const int tid = threadIdx.x;

    #pragma unroll
    for (int i = 0; i < N_ / THREADS; ++i) snm[tid + i * THREADS] = 1;
    __syncthreads();
    #pragma unroll
    for (int i = 0; i < 256 / THREADS; ++i) {
        int s = s_mti[tid + i * THREADS];
        snm[s & (N_ - 1)] = 0;
    }
    #pragma unroll
    for (int i = 0; i < (K_ * D_) / THREADS; ++i) {
        int e = tid + i * THREADS;
        int k = e >> 7, d = e & (D_ - 1);
        sWt[d * LDW + k]      = (_Float16)uP[e];
        sWt[d * LDW + K_ + k] = (_Float16)mP[e];
    }
    __syncthreads();

    const int lane = tid & 31;
    const int wave = tid >> 5;
    const int lhi  = lane >> 4;   // 0/1
    const int llo  = lane & 15;
    const int tstride = gridDim.x * WAVES;
    _Float16* mySlds = &sS[wave][0];

    for (int tile = blockIdx.x * WAVES + wave; tile < NTILES; tile += tstride) {
        const int    row0 = tile << 4;
        const float* X    = patches + (size_t)row0 * D_;
        float*       O    = out     + (size_t)row0 * D_;

        // ---- prefetch next tile's X (8 KB = 64 x 128B lines, 2 insts x 32 lanes) ----
        {
            int nt = tile + tstride;
            const float* Xn = patches + (size_t)((nt < NTILES) ? nt : tile) * 16 * D_;
            __builtin_prefetch(Xn + lane * 32, 0, 3);
            __builtin_prefetch(Xn + 1024 + lane * 32, 0, 3);
        }

        // ---- A1 fragments: 16x128 X tile, f32 -> f16 (WMMA 16-bit A layout) ----
        v16h a1[4];
        {
            const int    kb = lhi * 8;
            const float* xr = X + llo * D_;
            #pragma unroll
            for (int c = 0; c < 4; ++c) {
                const float* p0 = xr + c * 32 + kb;
                v4f x0 = *(const v4f*)(p0);
                v4f x1 = *(const v4f*)(p0 + 4);
                v4f x2 = *(const v4f*)(p0 + 16);
                v4f x3 = *(const v4f*)(p0 + 20);
                v16h a;
                #pragma unroll
                for (int q = 0; q < 4; ++q) {
                    a[q]      = (_Float16)x0[q];
                    a[4 + q]  = (_Float16)x1[q];
                    a[8 + q]  = (_Float16)x2[q];
                    a[12 + q] = (_Float16)x3[q];
                }
                a1[c] = a;
            }
        }

        // ---- per-row keep flags (C/D layout rows: m = lhi*8 + r) ----
        bool kb8[8];
        #pragma unroll
        for (int r = 0; r < 8; ++r) {
            int gm = row0 + lhi * 8 + r;
            unsigned q = (unsigned)gm / 25u;
            int p = gm - (int)q * 25;
            int n = (int)(q & (N_ - 1));
            kb8[r] = (p != T_MTI_) && snm[n];
        }

        // ---- GEMM1: S''[16x128] = X * [U;M]^T, sigmoid/threshold/mask -> LDS f16 ----
        #pragma unroll 2
        for (int j = 0; j < 8; ++j) {
            const float* Wg = (j < 4) ? uP : mP;   // stacked weight rows (L0/L2-resident)
            const int    jr = j & 3;
            v8f s = {};
            #pragma unroll
            for (int c = 0; c < 4; ++c) {
                // B frag: B[kk,n] = W''[j*16+n, c*32+kk]; lane n=llo, kk=lhi*8+h (h<8), 16+lhi*8+(h-8) (h>=8)
                const float* wr = Wg + (jr * 16 + llo) * D_ + c * 32 + lhi * 8;
                v16h b;
                #pragma unroll
                for (int q = 0; q < 16; q += 4) {
                    v4f w = *(const v4f*)(wr + q + (q & 8));
                    b[q]     = (_Float16)w[0];
                    b[q + 1] = (_Float16)w[1];
                    b[q + 2] = (_Float16)w[2];
                    b[q + 3] = (_Float16)w[3];
                }
                s = __builtin_amdgcn_wmma_f32_16x16x32_f16(
                        false, a1[c], false, b, (short)0, s, false, false);
            }
            #pragma unroll
            for (int r = 0; r < 8; ++r) {
                float sv   = s[r];
                bool  cond = (j < 4) ? kb8[r] : !kb8[r];        // U-cols: keep, M-cols: 1-keep
                float sig  = __builtin_amdgcn_rcpf(1.0f + __expf(-sv));
                float v    = (sv > 0.0f && cond) ? sig : 0.0f;  // sig>0.5 <=> sv>0
                mySlds[(lhi * 8 + r) * LDW + j * 16 + llo] = (_Float16)v;
            }
        }

        // ---- A2 fragments from S'' (LDS ops are in-order within a wave) ----
        v16h a2[4];
        {
            const int       kb = lhi * 8;
            const _Float16* sr = mySlds + llo * LDW;
            #pragma unroll
            for (int c = 0; c < 4; ++c) {
                union { v16h v; v8h h[2]; } u;
                u.h[0] = *(const v8h*)(sr + c * 32 + kb);
                u.h[1] = *(const v8h*)(sr + c * 32 + kb + 16);
                a2[c] = u.v;
            }
        }

        // ---- GEMM2: out = x + S'' * [U;M], one 16x16 d-tile at a time ----
        #pragma unroll 2
        for (int dt = 0; dt < 8; ++dt) {
            v8f acc;
            #pragma unroll
            for (int r = 0; r < 8; ++r)
                acc[r] = X[(lhi * 8 + r) * D_ + dt * 16 + llo];   // L0 hit: +x init
            #pragma unroll
            for (int c = 0; c < 4; ++c) {
                // B frag: B[kk,n] = W''[c*32+kk, dt*16+n] = sWt[(dt*16+n)*LDW + c*32+kk]
                const _Float16* wr = sWt + (dt * 16 + llo) * LDW + c * 32 + lhi * 8;
                union { v16h v; v8h h[2]; } u;
                u.h[0] = *(const v8h*)(wr);
                u.h[1] = *(const v8h*)(wr + 16);
                acc = __builtin_amdgcn_wmma_f32_16x16x32_f16(
                          false, a2[c], false, u.v, (short)0, acc, false, false);
            }
            // streaming result: non-temporal stores (write stream is never re-read)
            #pragma unroll
            for (int r = 0; r < 8; ++r)
                __builtin_nontemporal_store(acc[r], &O[(lhi * 8 + r) * D_ + dt * 16 + llo]);
        }
    }
}

extern "C" void kernel_launch(void* const* d_in, const int* in_sizes, int n_in,
                              void* d_out, int out_size, void* d_ws, size_t ws_size,
                              hipStream_t stream) {
    const float* patches = (const float*)d_in[0];
    const float* uP      = (const float*)d_in[1];
    const float* mP      = (const float*)d_in[2];
    const int*   s_mti   = (const int*)d_in[3];
    // d_in[4] (s_uti) is unused by the reference.
    float* out = (float*)d_out;

    // 25600 tiles, 4 waves/block -> 3200 blocks = exactly 2 tiles per wave
    // (second tile's X is prefetched during the first).
    dim3 grid(3200), block(THREADS);
    hipLaunchKernelGGL(ape_kernel, grid, block, 0, stream,
                       patches, uP, mP, s_mti, out);
}
